// Head_50843822850479
// MI455X (gfx1250) — matrix operand order repaired and checked by reading, not compiled, not run on hardware
//
#include <hip/hip_runtime.h>
#include <hip/hip_bf16.h>
#include <math.h>
#include <stdint.h>

// ---------- types ----------
typedef __bf16 bf16_t;
typedef bf16_t bf16x16 __attribute__((ext_vector_type(16)));
typedef bf16_t bf16x8  __attribute__((ext_vector_type(8)));
typedef float  f32x8   __attribute__((ext_vector_type(8)));
typedef float  f32x4   __attribute__((ext_vector_type(4)));

#define B_BATCH 8
#define T_SEQ   2048
#define C_DIM   1024
#define HD      128
#define BT      (B_BATCH * T_SEQ)         // 16384 rows
#define BT16    (BT / 16)                  // 1024 row-tiles
#define SCALE   0.03125f                   // C^-0.5 = 1/32

static __device__ inline f32x8 wmma_bf16(bf16x16 a, bf16x16 b, f32x8 c) {
    // D = A(16x32 bf16) * B(32x16 bf16) + C(16x16 f32)
    return __builtin_amdgcn_wmma_f32_16x16x32_bf16(
        /*neg_a=*/false, a, /*neg_b=*/false, b,
        /*c_mod=*/(short)0, c, /*reuse_a=*/false, /*reuse_b=*/false);
}

// gfx1250 async DMA: global -> LDS, 16 bytes per active lane, tracked by ASYNCcnt
static __device__ inline void async_b128(uint32_t lds_byte_addr, const void* gaddr) {
    asm volatile("global_load_async_to_lds_b128 %0, %1, off"
                 :: "v"(lds_byte_addr), "v"(gaddr) : "memory");
}
static __device__ inline void wait_async0() {
    asm volatile("s_wait_asynccnt 0" ::: "memory");
}
// generic (flat) address of an LDS object: low 32 bits are the LDS byte offset
static __device__ inline uint32_t lds_off(const void* p) {
    return (uint32_t)(uintptr_t)p;
}

// ------------------------------------------------------------------
// Kernel 1: W[c][h] f32  ->  Wt[h][c] bf16   (q,k,v via blockIdx.y)
// ------------------------------------------------------------------
__global__ __launch_bounds__(256)
void prep_weights(const float* __restrict__ Wq, const float* __restrict__ Wk,
                  const float* __restrict__ Wv, bf16_t* __restrict__ Wt) {
    int which = blockIdx.y;
    const float* W = (which == 0) ? Wq : (which == 1) ? Wk : Wv;
    bf16_t* dst = Wt + (size_t)which * HD * C_DIM;
    int idx = blockIdx.x * 256 + threadIdx.x;   // idx = h*1024 + c
    int h = idx >> 10;
    int c = idx & (C_DIM - 1);
    dst[idx] = (bf16_t)W[(size_t)c * HD + h];
}

// ------------------------------------------------------------------
// Kernel 2: QKV projection.  One wave -> one 16-row tile of one of
// {Q,K,V}.  A = x rows (f32 -> bf16 in regs), B = Wt rows (contiguous).
// ------------------------------------------------------------------
__global__ __launch_bounds__(128)
void qkv_proj(const float* __restrict__ x, const bf16_t* __restrict__ Wt,
              bf16_t* __restrict__ Qb, bf16_t* __restrict__ Kb,
              bf16_t* __restrict__ Vt) {
    const int wid  = threadIdx.x >> 5;
    const int ln   = threadIdx.x & 31;
    const int half = ln >> 4;          // 0: lanes 0-15, 1: lanes 16-31
    const int lc   = ln & 15;

    int wg = blockIdx.x * 4 + wid;     // 0 .. 3*BT16-1
    int which = wg / BT16;             // 0=q 1=k 2=v
    int tile  = wg - which * BT16;
    int t0    = tile * 16;             // flat row base (b*T + t)

    const bf16_t* Wsel = Wt + (size_t)which * HD * C_DIM;

    f32x8 acc[8];
    #pragma unroll
    for (int n = 0; n < 8; ++n) acc[n] = (f32x8){0,0,0,0,0,0,0,0};

    // A-matrix row for this lane (ISA layout: row = lane%16 in both halves)
    const float* xrow = x + (size_t)(t0 + lc) * C_DIM;

    for (int cb = 0; cb < C_DIM; cb += 32) {
        // A fragment 16x32: lane<16 holds K 0..15; lane>=16 K 16..31
        f32x4 a0 = *(const f32x4*)(xrow + cb + half * 16);
        f32x4 a1 = *(const f32x4*)(xrow + cb + half * 16 + 4);
        f32x4 a2 = *(const f32x4*)(xrow + cb + half * 16 + 8);
        f32x4 a3 = *(const f32x4*)(xrow + cb + half * 16 + 12);
        bf16x16 af;
        #pragma unroll
        for (int e = 0; e < 4; ++e) {
            af[e]      = (bf16_t)a0[e];
            af[e + 4]  = (bf16_t)a1[e];
            af[e + 8]  = (bf16_t)a2[e];
            af[e + 12] = (bf16_t)a3[e];
        }
        #pragma unroll
        for (int n = 0; n < 8; ++n) {
            // B fragment 32x16: lane = N(h), elements = K(c) with 16-split at half
            bf16x16 bfr = *(const bf16x16*)(Wsel + (size_t)(n * 16 + lc) * C_DIM
                                            + cb + half * 16);
            acc[n] = wmma_bf16(af, bfr, acc[n]);
        }
    }

    // C layout: VGPR r -> rows r (lanes 0-15) / r+8 (lanes 16-31); lane = N
    if (which < 2) {
        bf16_t* dst = which ? Kb : Qb;
        #pragma unroll
        for (int r = 0; r < 8; ++r) {
            int row = t0 + r + half * 8;
            #pragma unroll
            for (int n = 0; n < 8; ++n)
                dst[(size_t)row * HD + n * 16 + lc] = (bf16_t)acc[n][r];
        }
    } else {
        // store V transposed: Vt[b][h][t]
        #pragma unroll
        for (int r = 0; r < 8; ++r) {
            int row = t0 + r + half * 8;
            int bb = row / T_SEQ;
            int tt = row - bb * T_SEQ;
            #pragma unroll
            for (int n = 0; n < 8; ++n)
                Vt[(size_t)bb * HD * T_SEQ + (size_t)(n * 16 + lc) * T_SEQ + tt]
                    = (bf16_t)acc[n][r];
        }
    }
}

// ------------------------------------------------------------------
// Kernel 3: causal flash attention.
// One block = 64 query rows (4 waves x 16).  All waves iterate the same
// key-block count; K/V blocks (32 keys) are staged into LDS with
// double-buffered global_load_async_to_lds_b128 (ASYNCcnt) shared by
// the whole workgroup.  WMMA fragments then come from LDS.
// ------------------------------------------------------------------
__global__ __launch_bounds__(128)
void attn(const bf16_t* __restrict__ Qb, const bf16_t* __restrict__ Kb,
          const bf16_t* __restrict__ Vt, float* __restrict__ out) {
    __shared__ __align__(32) bf16_t Kl[2][32 * 128];   // [key][c]   8KB each
    __shared__ __align__(32) bf16_t Vl[2][128 * 32];   // [h][key]   8KB each
    __shared__ __align__(32) bf16_t Plds[4][16 * 32];  // per-wave P scratch

    const int tid  = threadIdx.x;
    const int wid  = tid >> 5;
    const int ln   = tid & 31;
    const int half = ln >> 4;
    const int lc   = ln & 15;

    const int Rblk = blockIdx.x * 64;        // flat query-row base of block
    const int b    = Rblk / T_SEQ;
    const int tqb  = Rblk - b * T_SEQ;       // in-batch base (multiple of 64)
    const int tq   = tqb + wid * 16;         // this wave's q tile
    const int R0   = Rblk + wid * 16;

    const bf16_t* kbase = Kb + (size_t)b * T_SEQ * HD;   // [key][c]
    const bf16_t* vbase = Vt + (size_t)b * HD * T_SEQ;   // [h][t]

    // ---- Q A-fragments (16 rows x 128 dims = 4 K-chunks), loaded once
    bf16x16 qa[4];
    const bf16_t* qrow = Qb + (size_t)(R0 + lc) * HD;
    #pragma unroll
    for (int c4 = 0; c4 < 4; ++c4) {
        int cb = c4 * 32;
        bf16x8 lo = *(const bf16x8*)(qrow + cb + half * 16);
        bf16x8 hi = *(const bf16x8*)(qrow + cb + half * 16 + 8);
        #pragma unroll
        for (int e = 0; e < 8; ++e) { qa[c4][e] = lo[e]; qa[c4][e + 8] = hi[e]; }
    }

    f32x8 o[8];
    #pragma unroll
    for (int n = 0; n < 8; ++n) o[n] = (f32x8){0,0,0,0,0,0,0,0};
    float m[8], l[8];
    #pragma unroll
    for (int r = 0; r < 8; ++r) { m[r] = -__builtin_inff(); l[r] = 0.0f; }

    // cooperative async stage of one 32-key block into buffer `buf`
    auto stage = [&](int k0, int buf) {
        // K block: 8KB contiguous (rows k0..k0+31, 256B each)
        const char* kg = (const char*)kbase + (size_t)k0 * HD * 2;
        uint32_t klds = lds_off(&Kl[buf][0]);
        #pragma unroll
        for (int j = 0; j < 4; ++j) {
            uint32_t off = (uint32_t)tid * 16 + j * 2048;
            async_b128(klds + off, kg + off);
        }
        // V block: 64B from each of 128 rows of Vt
        const char* vg = (const char*)vbase + (size_t)tid * T_SEQ * 2 + (size_t)k0 * 2;
        uint32_t vlds = lds_off(&Vl[buf][0]) + (uint32_t)tid * 64;
        #pragma unroll
        for (int j = 0; j < 4; ++j)
            async_b128(vlds + j * 16, vg + j * 16);
    };

    const int nblk = tqb / 32 + 2;          // keys 0 .. tqb+63 in 32-blocks

    stage(0, 0);
    wait_async0();
    __syncthreads();

    for (int i = 0; i < nblk; ++i) {
        const int k0  = i * 32;
        const int buf = i & 1;

        if (i + 1 < nblk) stage((i + 1) * 32, (i + 1) & 1);

        if (k0 <= tq + 15) {                 // causally active for this wave
            // ---- S = Q * K^T from LDS (keys k0..15 -> s0, k0+16..31 -> s1)
            f32x8 s0 = (f32x8){0,0,0,0,0,0,0,0};
            f32x8 s1 = (f32x8){0,0,0,0,0,0,0,0};
            #pragma unroll
            for (int c4 = 0; c4 < 4; ++c4) {
                int cb = c4 * 32;
                bf16x16 bk0 = *(const bf16x16*)(&Kl[buf][(lc)      * HD + cb + half * 16]);
                bf16x16 bk1 = *(const bf16x16*)(&Kl[buf][(16 + lc) * HD + cb + half * 16]);
                s0 = wmma_bf16(qa[c4], bk0, s0);
                s1 = wmma_bf16(qa[c4], bk1, s1);
            }

            const bool needmask = (k0 + 31 > tq);
            float p0[8], p1[8], corr[8];
            #pragma unroll
            for (int r = 0; r < 8; ++r) {
                int row = tq + r + half * 8;
                float v0 = s0[r] * SCALE;
                float v1 = s1[r] * SCALE;
                if (needmask) {
                    if (k0 + lc      > row) v0 = -__builtin_inff();
                    if (k0 + 16 + lc > row) v1 = -__builtin_inff();
                }
                float mx = fmaxf(v0, v1);
                #pragma unroll
                for (int off = 8; off; off >>= 1)
                    mx = fmaxf(mx, __shfl_xor(mx, off, 32));
                float mn = fmaxf(m[r], mx);
                float cr = __expf(m[r] - mn);
                float e0 = __expf(v0 - mn);
                float e1 = __expf(v1 - mn);
                float sum = e0 + e1;
                #pragma unroll
                for (int off = 8; off; off >>= 1)
                    sum += __shfl_xor(sum, off, 32);
                l[r] = l[r] * cr + sum;
                m[r] = mn;
                p0[r] = e0; p1[r] = e1; corr[r] = cr;
            }
            #pragma unroll
            for (int n = 0; n < 8; ++n)
                #pragma unroll
                for (int r = 0; r < 8; ++r) o[n][r] *= corr[r];

            // ---- P: C-layout -> A-layout through per-wave LDS scratch
            asm volatile("s_wait_dscnt 0" ::: "memory");
            #pragma unroll
            for (int r = 0; r < 8; ++r) {
                int row = r + half * 8;
                Plds[wid][row * 32 + lc]      = (bf16_t)p0[r];
                Plds[wid][row * 32 + 16 + lc] = (bf16_t)p1[r];
            }
            asm volatile("s_wait_dscnt 0" ::: "memory");
            bf16x16 pa;
            #pragma unroll
            for (int e = 0; e < 8; ++e) {
                pa[e]     = Plds[wid][lc * 32 + half * 16 + e];
                pa[e + 8] = Plds[wid][lc * 32 + half * 16 + 8 + e];
            }

            // ---- O += P(16x32) * V(32x16 per head-dim tile) from LDS
            #pragma unroll
            for (int n = 0; n < 8; ++n) {
                bf16x16 bv = *(const bf16x16*)(&Vl[buf][(n * 16 + lc) * 32 + half * 16]);
                o[n] = wmma_bf16(pa, bv, o[n]);
            }
        }

        wait_async0();                       // next buffer's DMA landed
        asm volatile("s_wait_dscnt 0" ::: "memory");
        __syncthreads();                     // all waves done reading buf
    }

    // epilogue: O /= l, f32 store
    #pragma unroll
    for (int r = 0; r < 8; ++r) {
        float inv = 1.0f / l[r];
        int row = R0 + r + half * 8;
        #pragma unroll
        for (int n = 0; n < 8; ++n)
            out[(size_t)row * HD + n * 16 + lc] = o[n][r] * inv;
    }
}

// ------------------------------------------------------------------
extern "C" void kernel_launch(void* const* d_in, const int* in_sizes, int n_in,
                              void* d_out, int out_size, void* d_ws, size_t ws_size,
                              hipStream_t stream) {
    const float* x  = (const float*)d_in[0];
    const float* Wq = (const float*)d_in[1];
    const float* Wk = (const float*)d_in[2];
    const float* Wv = (const float*)d_in[3];
    float* out = (float*)d_out;

    // workspace layout (bf16 elements): Wt[3][128][1024] | Q[BT][128] |
    // K[BT][128] | Vt[B][128][T]   -> ~12.75 MiB total
    bf16_t* wsb = (bf16_t*)d_ws;
    bf16_t* Wt = wsb;
    bf16_t* Qb = Wt + (size_t)3 * HD * C_DIM;
    bf16_t* Kb = Qb + (size_t)BT * HD;
    bf16_t* Vt = Kb + (size_t)BT * HD;

    prep_weights<<<dim3((HD * C_DIM) / 256, 3), 256, 0, stream>>>(Wq, Wk, Wv, Wt);
    qkv_proj<<<(3 * BT16) / 4, 128, 0, stream>>>(x, Wt, Qb, Kb, Vt);
    attn<<<BT / 64, 128, 0, stream>>>(Qb, Kb, Vt, out);
}
